// NAMTuringDecoder_5076651344121
// MI455X (gfx1250) — matrix operand order repaired and checked by reading, not compiled, not run on hardware
//
#include <hip/hip_runtime.h>
#include <hip/hip_bf16.h>
#include <stdint.h>

// ---------------------------------------------------------------------------
// Problem constants (from the reference)
// ---------------------------------------------------------------------------
#define DIM      256
#define N_TAPES  4
#define HEAD_DIM 64
#define HT       (HEAD_DIM * N_TAPES)       // 256
#define QKVA     (3 * HT + 11 * N_TAPES)    // 812
#define L_TGT    128
#define L_SRC    64
#define BATCH    512

// Fused GEMM: G[b, 0:1024]   = [h|x] @ [W_hh|W_ih]^T     (LSTM gates, pre-bias)
//             G[b, 1024:1836]= [h|x] @ [Wqkva|0]^T       (qkva, pre-bias)
#define NREAL  (4 * DIM + QKVA)   // 1836
#define NPAD   1856               // 29 * 64, multiple of 64 (block N tile)
#define KTOT   512                // DIM (h) + DIM (x)

typedef __attribute__((ext_vector_type(16))) __bf16 v16bf;
typedef __attribute__((ext_vector_type(8)))  float  v8f;

union ABFrag { uint4 u4[2]; v16bf v; };

__device__ __forceinline__ uint16_t f2bf(float f) {
    uint32_t u = __float_as_uint(f);
    uint32_t r = u + 0x7fffu + ((u >> 16) & 1u);   // round-to-nearest-even
    return (uint16_t)(r >> 16);
}
__device__ __forceinline__ float sigm(float x) { return 1.0f / (1.0f + __expf(-x)); }

// ---------------------------------------------------------------------------
// Setup kernel 1: pack [W_hh|W_ih ; Wqkva|0] -> bf16 W2[NPAD][KTOT], and the
// combined bias vector biasc[NPAD] (b_ih+b_hh for gates, Wqkva_b for qkva).
// ---------------------------------------------------------------------------
__global__ __launch_bounds__(256) void pack_weights_kernel(
    const float* __restrict__ W_ih, const float* __restrict__ W_hh,
    const float* __restrict__ b_ih, const float* __restrict__ b_hh,
    const float* __restrict__ Wq_w, const float* __restrict__ Wq_b,
    uint16_t* __restrict__ W2, float* __restrict__ biasc)
{
    int idx = blockIdx.x * blockDim.x + threadIdx.x;
    if (idx < NPAD * KTOT) {
        int n = idx >> 9;           // / 512
        int k = idx & 511;
        float w = 0.0f;
        if (n < 4 * DIM) {
            w = (k < DIM) ? W_hh[n * DIM + k] : W_ih[n * DIM + (k - DIM)];
        } else if (n < NREAL) {
            if (k < DIM) w = Wq_w[(n - 4 * DIM) * DIM + k];
        }
        W2[idx] = f2bf(w);
    }
    if (idx < NPAD) {
        float b = 0.0f;
        if (idx < 4 * DIM)      b = b_ih[idx] + b_hh[idx];
        else if (idx < NREAL)   b = Wq_b[idx - 4 * DIM];
        biasc[idx] = b;
    }
}

// ---------------------------------------------------------------------------
// Setup kernel 2: convert tgt -> bf16, copy src -> tape, zero h/c, init pos.
// ---------------------------------------------------------------------------
__global__ __launch_bounds__(256) void init_state_kernel(
    const float* __restrict__ tgt, const float* __restrict__ src,
    uint16_t* __restrict__ xbf, float* __restrict__ tape,
    uint16_t* __restrict__ hbf, float* __restrict__ cst,
    float* __restrict__ rpos, float* __restrict__ wpos)
{
    int idx = blockIdx.x * blockDim.x + threadIdx.x;
    const int NX = L_TGT * BATCH * DIM;          // 16,777,216
    const int NT = L_SRC * BATCH * DIM;          //  8,388,608
    const int NS = BATCH * DIM;                  //    131,072 (also L_SRC*BATCH*N_TAPES)
    if (idx < NX) xbf[idx] = f2bf(tgt[idx]);
    if (idx < NT) tape[idx] = src[idx];
    if (idx < NS) {
        hbf[idx] = 0;
        cst[idx] = 0.0f;
        int l = idx / (BATCH * N_TAPES);
        float p = (l == 0) ? 1.0f : 0.0f;
        rpos[idx] = p;
        wpos[idx] = p;
    }
}

// ---------------------------------------------------------------------------
// Per-step fused GEMM: G[512][NPAD] = [h|x_t] (bf16) @ W2^T (bf16), f32 acc.
// grid = (32, 29), block = 128 (4 waves). Each wave: one 16x16 tile, K-loop 16
// v_wmma_f32_16x16x32_bf16 ops. A tile staged in LDS.
// ---------------------------------------------------------------------------
__global__ __launch_bounds__(128) void gemm_step_kernel(
    const uint16_t* __restrict__ hbf, const uint16_t* __restrict__ xbf,
    const uint16_t* __restrict__ W2, float* __restrict__ G, int t)
{
    __shared__ uint16_t Als[16][KTOT + 8];   // pad 8 bf16 -> 1040B row stride

    const int tid  = threadIdx.x;
    const int m0   = blockIdx.x * 16;
    const int n0   = blockIdx.y * 64;

    // Cooperative A load: 16 rows x 512 bf16 = 1024 uint4 chunks of 8 bf16.
    for (int i = tid; i < 16 * 64; i += 128) {
        int row = i >> 6;
        int k   = (i & 63) * 8;
        const uint16_t* srcp = (k < DIM)
            ? (hbf + (size_t)(m0 + row) * DIM + k)
            : (xbf + ((size_t)t * BATCH + (m0 + row)) * DIM + (k - DIM));
        *(uint4*)&Als[row][k] = *(const uint4*)srcp;
    }
    __syncthreads();

    const int wave  = tid >> 5;
    const int lane  = tid & 31;
    const int khalf = lane >> 4;       // 0: K lower half per ISA layout, 1: upper
    const int arow  = lane & 15;       // A row (M) held by this lane
    const int ncol  = n0 + wave * 16 + (lane & 15);

    v8f acc = {};
    const uint16_t* brow = W2 + (size_t)ncol * KTOT;

    for (int k0 = 0; k0 < KTOT; k0 += 32) {
        // A fragment (16-bit A 16x32 layout): lane holds K = k0+khalf*8+{0..7}
        // in elems 0..7 and K = k0+16+khalf*8+{0..7} in elems 8..15.
        ABFrag a;
        a.u4[0] = *(const uint4*)&Als[arow][k0 + khalf * 8];
        a.u4[1] = *(const uint4*)&Als[arow][k0 + 16 + khalf * 8];
        // B fragment (32x16): same K order as A: elems 0..7 hold
        // K = k0+khalf*8+{0..7}, elems 8..15 hold K = k0+16+khalf*8+{0..7}.
        ABFrag b;
        const uint16_t* bp = brow + k0 + khalf * 8;
        b.u4[0] = *(const uint4*)bp;
        b.u4[1] = *(const uint4*)(bp + 16);
        __builtin_prefetch((const void*)(bp + 64), 0, 1);   // next K tile

        acc = __builtin_amdgcn_wmma_f32_16x16x32_bf16(
            false, a.v, false, b.v, (short)0, acc, false, false);
    }

    // C/D layout: VGPR r -> M = r + 8*(lane/16), N = lane%16.
    float* gout = G + (size_t)(m0 + 8 * khalf) * NPAD + ncol;
    #pragma unroll
    for (int r = 0; r < 8; ++r)
        gout[(size_t)r * NPAD] = acc[r];
}

// ---------------------------------------------------------------------------
// Per-step LSTM pointwise: gates = G[:,0:1024] + (b_ih+b_hh); c,h update.
// grid = 512 blocks x 256 threads -> one thread per (b, j).
// ---------------------------------------------------------------------------
__global__ __launch_bounds__(256) void lstm_step_kernel(
    const float* __restrict__ G, const float* __restrict__ biasc,
    float* __restrict__ cst, uint16_t* __restrict__ hbf)
{
    int idx = blockIdx.x * blockDim.x + threadIdx.x;   // b*256 + j
    int b = idx >> 8;
    int j = idx & 255;
    const float* Grow = G + (size_t)b * NPAD;

    float gi = Grow[          j] + biasc[          j];
    float gf = Grow[DIM     + j] + biasc[DIM     + j];
    float gg = Grow[2 * DIM + j] + biasc[2 * DIM + j];
    float go = Grow[3 * DIM + j] + biasc[3 * DIM + j];

    float c  = sigm(gf) * cst[idx] + sigm(gi) * tanhf(gg);
    float h  = sigm(go) * tanhf(c);
    cst[idx] = c;
    hbf[idx] = f2bf(h);
}

// ---------------------------------------------------------------------------
// Per-step tape kernel: one 64-thread block per (b, tape). Does query
// normalize, softmax dirs, sigmoid rwe, oldval/rval/jpos reductions, position
// roll-mix, in-place tape update, and the step's output row.
// ---------------------------------------------------------------------------
__global__ __launch_bounds__(64) void tape_step_kernel(
    const float* __restrict__ G, const float* __restrict__ biasc,
    float* __restrict__ tape, float* __restrict__ rpos, float* __restrict__ wpos,
    float* __restrict__ out, int step)
{
    __shared__ float tls[64 * 65];           // 64x64 tile, padded stride 65
    __shared__ float qsh[64], wls[64], rls[64], jls[64], red[64];
    __shared__ float ctrl[11];               // rd0-3, wd0-3, rwe0-2

    const int bt  = blockIdx.x;
    const int b   = bt >> 2;
    const int t   = bt & 3;
    const int tid = threadIdx.x;             // doubles as c (pass1/3) and l (pass2)
    const float* Grow = G + (size_t)b * NPAD;

    if (tid == 0) {
        float a[11];
        #pragma unroll
        for (int j = 0; j < 11; ++j)
            a[j] = Grow[4 * DIM + 3 * HT + t * 11 + j]
                 + biasc[4 * DIM + 3 * HT + t * 11 + j];
        float m1 = fmaxf(fmaxf(a[0], a[1]), fmaxf(a[2], a[3]));
        float e0 = __expf(a[0] - m1), e1 = __expf(a[1] - m1);
        float e2 = __expf(a[2] - m1), e3 = __expf(a[3] - m1);
        float s1 = e0 + e1 + e2 + e3;
        ctrl[0] = e0 / s1; ctrl[1] = e1 / s1; ctrl[2] = e2 / s1; ctrl[3] = e3 / s1;
        float m2 = fmaxf(fmaxf(a[4], a[5]), fmaxf(a[6], a[7]));
        float f0 = __expf(a[4] - m2), f1 = __expf(a[5] - m2);
        float f2 = __expf(a[6] - m2), f3 = __expf(a[7] - m2);
        float s2 = f0 + f1 + f2 + f3;
        ctrl[4] = f0 / s2; ctrl[5] = f1 / s2; ctrl[6] = f2 / s2; ctrl[7] = f3 / s2;
        ctrl[8] = sigm(a[8]); ctrl[9] = sigm(a[9]); ctrl[10] = sigm(a[10]);
    }

    const int posIdx = (tid * BATCH + b) * N_TAPES + t;   // [l][b][t]
    wls[tid] = wpos[posIdx];
    rls[tid] = rpos[posIdx];

    float q = Grow[4 * DIM +          t * HEAD_DIM + tid]
            + biasc[4 * DIM +          t * HEAD_DIM + tid];
    float v = Grow[4 * DIM + 2 * HT + t * HEAD_DIM + tid]
            + biasc[4 * DIM + 2 * HT + t * HEAD_DIM + tid];

    red[tid] = q * q;
    __syncthreads();
    for (int s = 32; s > 0; s >>= 1) {
        if (tid < s) red[tid] += red[tid + s];
        __syncthreads();
    }
    qsh[tid] = q / fmaxf(sqrtf(red[0]), 1e-12f);
    __syncthreads();

    // Pass 1 (thread = c): stage tile, oldval & rval reductions over L.
    float oldv = 0.0f, racc = 0.0f;
    const int c = tid;
    for (int l = 0; l < 64; ++l) {
        float tv = tape[((size_t)(l * BATCH + b) * N_TAPES + t) * HEAD_DIM + c];
        tls[l * 65 + c] = tv;
        oldv += tv * wls[l];
        racc += tv * rls[l];
    }
    __syncthreads();

    // Pass 2 (thread = l): jpos = normalize_L( tape . query ).
    {
        const int l = tid;
        float jq = 0.0f;
        for (int cc = 0; cc < 64; ++cc) jq += tls[l * 65 + cc] * qsh[cc];
        red[tid] = jq * jq;
        __syncthreads();
        for (int s = 32; s > 0; s >>= 1) {
            if (tid < s) red[tid] += red[tid + s];
            __syncthreads();
        }
        jls[l] = jq / fmaxf(sqrtf(red[0]), 1e-12f);
    }
    __syncthreads();

    // Position updates (thread = l): roll(+1)[l]=pos[l-1], roll(-1)[l]=pos[l+1].
    {
        const int l  = tid;
        const int lp = (l + 1) & 63;       // prev_* = roll(-1)
        const int lm = (l + 63) & 63;      // next_* = roll(+1)
        float nw = rls[0]; (void)nw;
        float w_ = wls[lp] * ctrl[4] + wls[l] * ctrl[5] + wls[lm] * ctrl[6] + jls[l] * ctrl[7];
        float r_ = rls[lp] * ctrl[0] + rls[l] * ctrl[1] + rls[lm] * ctrl[2] + jls[l] * ctrl[3];
        wpos[posIdx] = w_;
        rpos[posIdx] = r_;
    }

    // Pass 3 (thread = c): tape += wpos * (value*rwe1 - oldval*rwe2); output.
    const float delta = v * ctrl[9] - oldv * ctrl[10];
    for (int l = 0; l < 64; ++l) {
        size_t gi = ((size_t)(l * BATCH + b) * N_TAPES + t) * HEAD_DIM + c;
        tape[gi] = tls[l * 65 + c] + wls[l] * delta;
    }
    out[((size_t)step * BATCH + b) * DIM + t * HEAD_DIM + c] = racc * ctrl[8];
}

// ---------------------------------------------------------------------------
// Host: workspace layout + launch sequence (all on `stream`, capture-safe).
// ---------------------------------------------------------------------------
static inline size_t alignUp(size_t x, size_t a) { return (x + a - 1) & ~(a - 1); }

extern "C" void kernel_launch(void* const* d_in, const int* in_sizes, int n_in,
                              void* d_out, int out_size, void* d_ws, size_t ws_size,
                              hipStream_t stream) {
    (void)in_sizes; (void)n_in; (void)out_size; (void)ws_size;
    const float* tgt   = (const float*)d_in[0];
    const float* src   = (const float*)d_in[1];
    const float* W_ih  = (const float*)d_in[2];
    const float* W_hh  = (const float*)d_in[3];
    const float* b_ih  = (const float*)d_in[4];
    const float* b_hh  = (const float*)d_in[5];
    const float* Wq_w  = (const float*)d_in[6];
    const float* Wq_b  = (const float*)d_in[7];
    float* out = (float*)d_out;

    // Workspace carve-up (~74 MB total).
    char* ws = (char*)d_ws;
    size_t off = 0;
    uint16_t* W2    = (uint16_t*)(ws + off); off = alignUp(off + (size_t)NPAD * KTOT * 2, 256);
    float*    biasc = (float*)   (ws + off); off = alignUp(off + (size_t)NPAD * 4, 256);
    uint16_t* hbf   = (uint16_t*)(ws + off); off = alignUp(off + (size_t)BATCH * DIM * 2, 256);
    float*    cst   = (float*)   (ws + off); off = alignUp(off + (size_t)BATCH * DIM * 4, 256);
    uint16_t* xbf   = (uint16_t*)(ws + off); off = alignUp(off + (size_t)L_TGT * BATCH * DIM * 2, 256);
    float*    tape  = (float*)   (ws + off); off = alignUp(off + (size_t)L_SRC * BATCH * DIM * 4, 256);
    float*    rpos  = (float*)   (ws + off); off = alignUp(off + (size_t)L_SRC * BATCH * N_TAPES * 4, 256);
    float*    wpos  = (float*)   (ws + off); off = alignUp(off + (size_t)L_SRC * BATCH * N_TAPES * 4, 256);
    float*    G     = (float*)   (ws + off); off = alignUp(off + (size_t)BATCH * NPAD * 4, 256);

    // Setup: pack weights (bf16) + biases, init state (x bf16, tape, h, c, pos).
    {
        int n = NPAD * KTOT;
        pack_weights_kernel<<<(n + 255) / 256, 256, 0, stream>>>(
            W_ih, W_hh, b_ih, b_hh, Wq_w, Wq_b, W2, biasc);
        int m = L_TGT * BATCH * DIM;
        init_state_kernel<<<(m + 255) / 256, 256, 0, stream>>>(
            tgt, src, xbf, tape, hbf, cst, rpos, wpos);
    }

    // 128 recurrent steps: fused WMMA GEMM -> LSTM pointwise -> tape update.
    dim3 gemmGrid(BATCH / 16, NPAD / 64);
    for (int t = 0; t < L_TGT; ++t) {
        gemm_step_kernel<<<gemmGrid, 128, 0, stream>>>(hbf, xbf, W2, G, t);
        lstm_step_kernel<<<BATCH, DIM, 0, stream>>>(G, biasc, cst, hbf);
        tape_step_kernel<<<BATCH * N_TAPES, 64, 0, stream>>>(
            G, biasc, tape, rpos, wpos, out, t);
    }
}
